// MultiHeadAttention_47253230191317
// MI455X (gfx1250) — matrix operand order repaired and checked by reading, not compiled, not run on hardware
//
#include <hip/hip_runtime.h>
#include <hip/hip_bf16.h>
#include <math.h>

// ---------------------------------------------------------------------------
// MI455X (gfx1250, wave32) shape-biased multi-head attention.
// All GEMM stages run on V_WMMA_F32_16X16X32_BF16 (fp32 accumulate, bf16 in).
// Fragment loads are 128-bit vectorized; V is stored transposed so the
// P@V B-fragments are contiguous; projection GEMMs amortize each strided
// weight B-fragment across 4 row-tiles (4 WMMAs per load).
// ---------------------------------------------------------------------------

typedef __attribute__((ext_vector_type(16))) __bf16 v16bf;
typedef __attribute__((ext_vector_type(8)))  __bf16 v8bf;
typedef __attribute__((ext_vector_type(4)))  __bf16 v4bf;
typedef __attribute__((ext_vector_type(8)))  float  v8f;
typedef __attribute__((ext_vector_type(4)))  float  v4f;

#define BATCH   16
#define SEQ     577
#define SEQP    608         // 577 padded to j-loop coverage (19*32)
#define PATCH   576
#define DM      768
#define NH      12
#define DH      64
#define NT      37          // ceil(577/16) row tiles
#define RG      10          // ceil(37/4)  row-tile groups of 4
#define CT      48          // 768/16 column tiles
#define INV_TEMP 0.125f     // 1/sqrt(64)

// ---- workspace layout (bytes) ----
#define SZ_HEAD   ((size_t)BATCH*NH*SEQ*DH*2)      // bf16 (B,H,N,64)
#define SZ_VT     ((size_t)BATCH*NH*DH*SEQP*2)     // bf16 (B,H,64,608) transposed V
#define SZ_ENORM  ((size_t)BATCH*PATCH*DM*2)       // bf16
#define SZ_PEN    ((size_t)BATCH*PATCH*PATCH*4)    // f32
#define SZ_ATTN   ((size_t)BATCH*SEQ*DM*2)         // bf16
#define OFF_QH    ((size_t)0)
#define OFF_KH    (OFF_QH + SZ_HEAD)
#define OFF_VT    (OFF_KH + SZ_HEAD)
#define OFF_EN    (OFF_VT + SZ_VT)
#define OFF_PEN   (OFF_EN + SZ_ENORM)
#define OFF_AO    (OFF_PEN + SZ_PEN)
#define OFF_FC    (OFF_AO + SZ_ATTN)

static __device__ __forceinline__ v8f wmma_bf16(v16bf a, v16bf b, v8f c) {
    return __builtin_amdgcn_wmma_f32_16x16x32_bf16(false, a, false, b,
                                                   (short)0, c, false, false);
}

// ---- vectorized fragment loaders (all call sites are 16B aligned) ----

// A fragment (16x32): lane&15 = row m, hi = lane>>4 selects K slice
// {hi*16..hi*16+15}.  Two b128 loads.
static __device__ __forceinline__ v16bf load_a_bf16v(const __bf16* row, int k0, int hi) {
    const v8bf* p = (const v8bf*)(row + k0 + hi * 16);
    v8bf x = p[0], y = p[1];
    v16bf a;
#pragma unroll
    for (int i = 0; i < 8; ++i) { a[i] = x[i]; a[i + 8] = y[i]; }
    return a;
}
// Same slice from an fp32 row: four b128 loads + pack-convert.
static __device__ __forceinline__ v16bf load_a_f32v(const float* row, int k0, int hi) {
    const v4f* p = (const v4f*)(row + k0 + hi * 16);
    v4f x0 = p[0], x1 = p[1], y0 = p[2], y1 = p[3];
    v16bf a;
#pragma unroll
    for (int i = 0; i < 4; ++i) {
        a[i]      = (__bf16)x0[i];  a[i + 4]  = (__bf16)x1[i];
        a[i + 8]  = (__bf16)y0[i];  a[i + 12] = (__bf16)y1[i];
    }
    return a;
}
// 16 contiguous bf16 (one B-fragment lane column): two b128 loads.
static __device__ __forceinline__ v16bf load_b16_contig(const __bf16* p) {
    const v8bf* q = (const v8bf*)p;
    v8bf x = q[0], y = q[1];
    v16bf r;
#pragma unroll
    for (int i = 0; i < 8; ++i) { r[i] = x[i]; r[i + 8] = y[i]; }
    return r;
}
// B fragment from fp32 row-major W: column = lane&15, K strided by ldb.
static __device__ __forceinline__ v16bf load_b_f32_strided(const float* W, int ldb,
                                                           int k0, int col, int hi) {
    v16bf b;
    const float* p = W + (size_t)(k0 + hi * 16) * ldb + col;
#pragma unroll
    for (int i = 0; i < 16; ++i) b[i] = (__bf16)p[(size_t)i * ldb];
    return b;
}

// ===========================================================================
// 1. QKV projection: each wave computes a 64x16 output slab (4 row-tiles
//    share one strided weight B-fragment per k-step -> 4 WMMAs per B load).
//    Q scaled by 1/sqrt(dk).  V stored transposed (B,H,64,608).
// ===========================================================================
__global__ void qkv_proj_kernel(const float* __restrict__ q, const float* __restrict__ k,
                                const float* __restrict__ v,
                                const float* __restrict__ wq, const float* __restrict__ wk,
                                const float* __restrict__ wv,
                                __bf16* __restrict__ qh, __bf16* __restrict__ kh,
                                __bf16* __restrict__ vt) {
    int task = blockIdx.x * (blockDim.x >> 5) + (threadIdx.x >> 5);
    const int per = BATCH * RG * CT;
    if (task >= 3 * per) return;
    int which = task / per;           task -= which * per;
    int b     = task / (RG * CT);     task -= b * (RG * CT);
    int rg    = task / CT;
    int ct    = task - rg * CT;

    const float* X = (which == 0) ? q  : (which == 1) ? k  : v;
    const float* W = (which == 0) ? wq : (which == 1) ? wk : wv;
    const float scale = (which == 0) ? INV_TEMP : 1.0f;

    const int lane = threadIdx.x & 31;
    const int m = lane & 15, hi = lane >> 4;
    const int r0 = rg * 64, c0 = ct * 16;
    const int h = ct >> 2, dbase = (ct & 3) * 16;

    const float* arow[4];
#pragma unroll
    for (int s = 0; s < 4; ++s)
        arow[s] = X + ((size_t)b * SEQ + min(r0 + s * 16 + m, SEQ - 1)) * DM;

    v8f zero = {};
    v8f acc[4];
#pragma unroll
    for (int s = 0; s < 4; ++s) acc[s] = zero;

    for (int k0 = 0; k0 < DM; k0 += 32) {
        v16bf bf = load_b_f32_strided(W, DM, k0, c0 + m, hi);
#pragma unroll
        for (int s = 0; s < 4; ++s) {
            v16bf af = load_a_f32v(arow[s], k0, hi);
            acc[s] = wmma_bf16(af, bf, acc[s]);
        }
    }
    if (which == 2) {                 // V -> transposed (d-major, row contiguous)
        __bf16* O = vt + ((size_t)(b * NH + h) * DH + dbase + m) * SEQP;
#pragma unroll
        for (int s = 0; s < 4; ++s)
#pragma unroll
            for (int r = 0; r < 8; ++r) {
                int row = r0 + s * 16 + hi * 8 + r;
                if (row < SEQ) O[row] = (__bf16)acc[s][r];
            }
    } else {
        __bf16* O = ((which == 0) ? qh : kh) + (size_t)(b * NH + h) * SEQ * DH;
#pragma unroll
        for (int s = 0; s < 4; ++s)
#pragma unroll
            for (int r = 0; r < 8; ++r) {
                int row = r0 + s * 16 + hi * 8 + r;
                if (row < SEQ)
                    O[(size_t)row * DH + dbase + m] = (__bf16)(acc[s][r] * scale);
            }
    }
}

// ===========================================================================
// 2. Normalize patch embeddings (skip CLS) -> bf16, float4 vectorized.
// ===========================================================================
__global__ void norm_emb_kernel(const float* __restrict__ emb, __bf16* __restrict__ en) {
    int task = blockIdx.x * (blockDim.x >> 5) + (threadIdx.x >> 5);
    if (task >= BATCH * PATCH) return;
    int b = task / PATCH, i = task - b * PATCH;
    const float* src = emb + ((size_t)b * SEQ + i + 1) * DM;
    const int lane = threadIdx.x & 31;
    v4f vals[6];
    float ss = 0.f;
#pragma unroll
    for (int t = 0; t < 6; ++t) {
        vals[t] = ((const v4f*)src)[lane + t * 32];
#pragma unroll
        for (int e = 0; e < 4; ++e) ss += vals[t][e] * vals[t][e];
    }
#pragma unroll
    for (int msk = 1; msk < 32; msk <<= 1) ss += __shfl_xor(ss, msk, 32);
    float sc = rsqrtf(ss + 1e-12f);
    v4bf* dst = (v4bf*)(en + (size_t)task * DM);
#pragma unroll
    for (int t = 0; t < 6; ++t) {
        v4bf o;
#pragma unroll
        for (int e = 0; e < 4; ++e) o[e] = (__bf16)(vals[t][e] * sc);
        dst[lane + t * 32] = o;
    }
}

// ===========================================================================
// 3. Penalty: alpha * (e_i . e_j) * dist_scale * ||p_i - p_j|| via bf16 WMMA.
// ===========================================================================
__global__ void penalty_kernel(const __bf16* __restrict__ en,
                               const float* __restrict__ pos,
                               float* __restrict__ pen) {
    int task = blockIdx.x * (blockDim.x >> 5) + (threadIdx.x >> 5);
    const int TPB = (PATCH / 16) * (PATCH / 16);   // 36*36
    if (task >= BATCH * TPB) return;
    int b  = task / TPB;           task -= b * TPB;
    int it = task / (PATCH / 16);
    int jt = task - it * (PATCH / 16);
    const int lane = threadIdx.x & 31;
    const int m = lane & 15, hi = lane >> 4;
    const int i0 = it * 16, j0 = jt * 16;

    const __bf16* Ea = en + ((size_t)b * PATCH + i0 + m) * DM;
    const __bf16* Eb = en + ((size_t)b * PATCH + j0 + m) * DM;
    v8f acc = {};
    for (int k0 = 0; k0 < DM; k0 += 32) {
        v16bf af = load_a_bf16v(Ea, k0, hi);
        v16bf bf = load_b16_contig(Eb + k0 + hi * 16);
        acc = wmma_bf16(af, bf, acc);
    }
    const float* pb = pos + (size_t)b * PATCH * 2;
    int j = j0 + m;
    float pjx = pb[j * 2], pjy = pb[j * 2 + 1];
    float* pp = pen + (size_t)b * PATCH * PATCH;
#pragma unroll
    for (int r = 0; r < 8; ++r) {
        int i = i0 + hi * 8 + r;
        float dx = pb[i * 2] - pjx, dy = pb[i * 2 + 1] - pjy;
        float dist = sqrtf(dx * dx + dy * dy + 1e-12f);
        pp[(size_t)i * PATCH + j] = acc[r] * dist;   // ALPHA = DIST_SCALE = 1
    }
}

// ===========================================================================
// 4. Flash attention with penalty: per (b,h,16-row tile), online softmax.
//    V^T pad columns [577,608) are zero, masked probabilities are exact 0.
// ===========================================================================
__global__ void attn_kernel(const __bf16* __restrict__ qh, const __bf16* __restrict__ kh,
                            const __bf16* __restrict__ vt, const float* __restrict__ pen,
                            __bf16* __restrict__ ao) {
    __shared__ __bf16 plds[4][16 * 32];
    int wave = threadIdx.x >> 5;
    int task = blockIdx.x * (blockDim.x >> 5) + wave;
    if (task >= BATCH * NH * NT) return;
    int b  = task / (NH * NT);  task -= b * (NH * NT);
    int h  = task / NT;
    int it = task - h * NT;
    const int lane = threadIdx.x & 31;
    const int m = lane & 15, hi = lane >> 4;
    const int i0 = it * 16;

    const __bf16* Q  = qh + (size_t)(b * NH + h) * SEQ * DH;
    const __bf16* K  = kh + (size_t)(b * NH + h) * SEQ * DH;
    const __bf16* VT = vt + (size_t)(b * NH + h) * DH * SEQP;
    const float*  P  = pen + (size_t)b * PATCH * PATCH;
    __bf16* pb = plds[wave];

    // Q fragments stay resident across the j loop (K = 0..31 and 32..63).
    const __bf16* qrow = Q + (size_t)min(i0 + m, SEQ - 1) * DH;
    v16bf aq0 = load_a_bf16v(qrow, 0, hi);
    v16bf aq1 = load_a_bf16v(qrow, 32, hi);

    float rmax[8], rsum[8];
    v8f o[4];
    v8f zero = {};
#pragma unroll
    for (int r = 0; r < 8; ++r) { rmax[r] = -1e30f; rsum[r] = 0.f; }
#pragma unroll
    for (int t = 0; t < 4; ++t) o[t] = zero;

    for (int j0 = 0; j0 < SEQP; j0 += 32) {     // 19 chunks cover 577 (+mask)
        if (j0 + 32 < SEQ)                       // global_prefetch next K chunk
            __builtin_prefetch(K + (size_t)(j0 + 32 + m) * DH, 0, 1);

        // ---- S = Q @ K^T for this 16x32 chunk ----
        v8f s[2];
#pragma unroll
        for (int half = 0; half < 2; ++half) {
            int jc = min(j0 + half * 16 + m, SEQ - 1);
            const __bf16* kr = K + (size_t)jc * DH;
            v16bf bk = load_b16_contig(kr + hi * 16);
            v8f sc = wmma_bf16(aq0, bk, zero);
            bk = load_b16_contig(kr + 32 + hi * 16);
            s[half] = wmma_bf16(aq1, bk, sc);
        }
        // ---- penalty + mask (C layout: row = hi*8+r, col = lane&15) ----
#pragma unroll
        for (int half = 0; half < 2; ++half) {
#pragma unroll
            for (int r = 0; r < 8; ++r) {
                int i = i0 + hi * 8 + r;
                int j = j0 + half * 16 + m;
                float sv = s[half][r];
                if (j >= SEQ) sv = -1e30f;
                else if (i >= 1 && i < SEQ && j >= 1)
                    sv -= P[(size_t)(i - 1) * PATCH + (j - 1)];
                s[half][r] = sv;
            }
        }
        // ---- online softmax (half-wave reductions over 16 columns) ----
        float p0[8], p1[8];
#pragma unroll
        for (int r = 0; r < 8; ++r) {
            float mx = fmaxf(s[0][r], s[1][r]);
#pragma unroll
            for (int msk = 1; msk < 16; msk <<= 1) mx = fmaxf(mx, __shfl_xor(mx, msk, 32));
            float nm = fmaxf(rmax[r], mx);
            float corr = __expf(rmax[r] - nm);
            rmax[r] = nm;
            p0[r] = __expf(s[0][r] - nm);
            p1[r] = __expf(s[1][r] - nm);
            float ps = p0[r] + p1[r];
#pragma unroll
            for (int msk = 1; msk < 16; msk <<= 1) ps += __shfl_xor(ps, msk, 32);
            rsum[r] = rsum[r] * corr + ps;
#pragma unroll
            for (int t = 0; t < 4; ++t) o[t][r] *= corr;
        }
        // ---- re-fragment P: C layout -> LDS -> A layout ----
#pragma unroll
        for (int r = 0; r < 8; ++r) {
            int mrow = hi * 8 + r;
            pb[mrow * 32 + m]      = (__bf16)p0[r];
            pb[mrow * 32 + 16 + m] = (__bf16)p1[r];
        }
        v16bf ap = load_a_bf16v(pb + m * 32, 0, hi);
        // ---- O += P @ V : V^T rows are contiguous b128 loads ----
#pragma unroll
        for (int t = 0; t < 4; ++t) {
            v16bf bv = load_b16_contig(VT + (size_t)(t * 16 + m) * SEQP + j0 + hi * 16);
            o[t] = wmma_bf16(ap, bv, o[t]);
        }
    }
    // ---- epilogue: normalize and emit bf16 (B,N,H*64) ----
    __bf16* AO = ao + (size_t)b * SEQ * DM;
#pragma unroll
    for (int r = 0; r < 8; ++r) {
        int row = i0 + hi * 8 + r;
        if (row < SEQ) {
            float inv = 1.0f / rsum[r];
#pragma unroll
            for (int t = 0; t < 4; ++t)
                AO[(size_t)row * DM + h * DH + t * 16 + m] = (__bf16)(o[t][r] * inv);
        }
    }
}

// ===========================================================================
// 5. Output projection + residual: 64x16 slab per wave (B-fragment reuse).
// ===========================================================================
__global__ void fc_kernel(const __bf16* __restrict__ ao, const float* __restrict__ wfc,
                          const float* __restrict__ qres, float* __restrict__ fc) {
    int task = blockIdx.x * (blockDim.x >> 5) + (threadIdx.x >> 5);
    if (task >= BATCH * RG * CT) return;
    int b  = task / (RG * CT);   task -= b * (RG * CT);
    int rg = task / CT;
    int ct = task - rg * CT;
    const int lane = threadIdx.x & 31;
    const int m = lane & 15, hi = lane >> 4;
    const int r0 = rg * 64, col = ct * 16 + m;

    const __bf16* arow[4];
#pragma unroll
    for (int s = 0; s < 4; ++s)
        arow[s] = ao + ((size_t)b * SEQ + min(r0 + s * 16 + m, SEQ - 1)) * DM;

    v8f zero = {};
    v8f acc[4];
#pragma unroll
    for (int s = 0; s < 4; ++s) acc[s] = zero;

    for (int k0 = 0; k0 < DM; k0 += 32) {
        v16bf bf = load_b_f32_strided(wfc, DM, k0, col, hi);
#pragma unroll
        for (int s = 0; s < 4; ++s) {
            v16bf af = load_a_bf16v(arow[s], k0, hi);
            acc[s] = wmma_bf16(af, bf, acc[s]);
        }
    }
#pragma unroll
    for (int s = 0; s < 4; ++s)
#pragma unroll
        for (int r = 0; r < 8; ++r) {
            int row = r0 + s * 16 + hi * 8 + r;
            if (row < SEQ) {
                size_t g = ((size_t)b * SEQ + row) * DM + col;
                fc[g] = acc[s][r] + qres[g];
            }
        }
}

// ===========================================================================
// 6. LayerNorm -> d_out (fp32), float4 vectorized.
// ===========================================================================
__global__ void ln_kernel(const float* __restrict__ fc, const float* __restrict__ gamma,
                          const float* __restrict__ beta, float* __restrict__ out) {
    int row = blockIdx.x * (blockDim.x >> 5) + (threadIdx.x >> 5);
    if (row >= BATCH * SEQ) return;
    const int lane = threadIdx.x & 31;
    const v4f* src = (const v4f*)(fc + (size_t)row * DM);
    v4f v[6];
    float s = 0.f;
#pragma unroll
    for (int t = 0; t < 6; ++t) {
        v[t] = src[lane + t * 32];
#pragma unroll
        for (int e = 0; e < 4; ++e) s += v[t][e];
    }
#pragma unroll
    for (int msk = 1; msk < 32; msk <<= 1) s += __shfl_xor(s, msk, 32);
    float mu = s * (1.0f / DM);
    float var = 0.f;
#pragma unroll
    for (int t = 0; t < 6; ++t)
#pragma unroll
        for (int e = 0; e < 4; ++e) { float d = v[t][e] - mu; var += d * d; }
#pragma unroll
    for (int msk = 1; msk < 32; msk <<= 1) var += __shfl_xor(var, msk, 32);
    float inv = rsqrtf(var * (1.0f / DM) + 1e-6f);
    v4f* dst = (v4f*)(out + (size_t)row * DM);
    const v4f* g4 = (const v4f*)gamma;
    const v4f* b4 = (const v4f*)beta;
#pragma unroll
    for (int t = 0; t < 6; ++t) {
        v4f gg = g4[lane + t * 32], bb = b4[lane + t * 32], oo;
#pragma unroll
        for (int e = 0; e < 4; ++e) oo[e] = (v[t][e] - mu) * inv * gg[e] + bb[e];
        dst[lane + t * 32] = oo;
    }
}

// ===========================================================================
extern "C" void kernel_launch(void* const* d_in, const int* in_sizes, int n_in,
                              void* d_out, int out_size, void* d_ws, size_t ws_size,
                              hipStream_t stream) {
    const float* q    = (const float*)d_in[0];
    const float* k    = (const float*)d_in[1];
    const float* v    = (const float*)d_in[2];
    const float* pos  = (const float*)d_in[3];
    const float* emb  = (const float*)d_in[4];
    const float* wqs  = (const float*)d_in[5];
    const float* wks  = (const float*)d_in[6];
    const float* wvs  = (const float*)d_in[7];
    const float* wfc  = (const float*)d_in[8];
    const float* lng  = (const float*)d_in[9];
    const float* lnb  = (const float*)d_in[10];
    float* out = (float*)d_out;

    char* ws = (char*)d_ws;
    __bf16* qh = (__bf16*)(ws + OFF_QH);
    __bf16* kh = (__bf16*)(ws + OFF_KH);
    __bf16* vt = (__bf16*)(ws + OFF_VT);
    __bf16* en = (__bf16*)(ws + OFF_EN);
    float*  pn = (float*)(ws + OFF_PEN);
    __bf16* ao = (__bf16*)(ws + OFF_AO);
    float*  fc = (float*)(ws + OFF_FC);

    const int WPB = 4, TPB = 128;   // 4 waves / block

    // zero V^T (covers pad columns [577,608) so masked j contributes exact 0)
    hipMemsetAsync(vt, 0, SZ_VT, stream);

    int t1 = 3 * BATCH * RG * CT;                       // 23040 waves
    qkv_proj_kernel<<<(t1 + WPB - 1) / WPB, TPB, 0, stream>>>(q, k, v, wqs, wks, wvs,
                                                              qh, kh, vt);
    int t2 = BATCH * PATCH;                             // 9216 waves
    norm_emb_kernel<<<(t2 + WPB - 1) / WPB, TPB, 0, stream>>>(emb, en);

    int t3 = BATCH * (PATCH / 16) * (PATCH / 16);       // 20736 waves
    penalty_kernel<<<(t3 + WPB - 1) / WPB, TPB, 0, stream>>>(en, pos, pn);

    int t4 = BATCH * NH * NT;                           // 7104 waves
    attn_kernel<<<(t4 + WPB - 1) / WPB, TPB, 0, stream>>>(qh, kh, vt, pn, ao);

    int t5 = BATCH * RG * CT;                           // 7680 waves
    fc_kernel<<<(t5 + WPB - 1) / WPB, TPB, 0, stream>>>(ao, wfc, q, fc);

    int t6 = BATCH * SEQ;                               // 9232 waves
    ln_kernel<<<(t6 + WPB - 1) / WPB, TPB, 0, stream>>>(fc, lng, lnb, out);
}
